// LinearAttention_54760833024195
// MI455X (gfx1250) — compile-verified, matrix operand order checked by reading, not run on hardware
//
#include <hip/hip_runtime.h>

// ---- problem constants (match reference) ----
#define NSEQ 4096
#define BB   8
#define CC   512
#define HH   8
#define DD   64
#define BH   (BB*HH)        // 64 (b,h) pairs
#define MTOT (NSEQ*BB)      // 32768 rows for the big GEMMs
#define EPSF 1e-6f

typedef __attribute__((ext_vector_type(16))) _Float16 v16h;
typedef __attribute__((ext_vector_type(8)))  _Float16 v8h;
typedef __attribute__((ext_vector_type(4)))  _Float16 v4h;
typedef __attribute__((ext_vector_type(8)))  float    v8f;
typedef __attribute__((ext_vector_type(4)))  float    v4f;

// fast sigmoid / reciprocal: v_rcp_f32 instead of IEEE div expansion
__device__ __forceinline__ float fast_rcp(float x){ return __builtin_amdgcn_rcpf(x); }
__device__ __forceinline__ float sigmoidf(float x){ return fast_rcp(1.0f + __expf(-x)); }

// WMMA f16 16x16x32 register layout (ISA 7.12.2, wave32):
//   lanes 0-15  hold K = 0..7  (elems 0-7) and K = 16..23 (elems 8-15)
//   lanes 16-31 hold K = 8..15 (elems 0-7) and K = 24..31 (elems 8-15)
// For A: "row" is the M index; for B: "row" is the N column index.
__device__ __forceinline__ void frag_pos(int k, int ncol, int& lane, int& elem){
  int hi = (k >> 3) & 1;
  lane = (ncol & 15) + hi*16;
  elem = (k & 7) + ((k & 16) ? 8 : 0);
}

// A-fragment (16x32 f16) from a row-major tile, row stride ld (elements).
// Lowers to 2x b128 loads per lane (contiguous 16B chunks).
__device__ __forceinline__ v16h load_frag_a(const _Float16* src, int ld){
  int lane = threadIdx.x & 31;
  int r  = lane & 15;
  int kb = (lane & 16) ? 8 : 0;
  const _Float16* p = src + r*ld;
  v16h a;
#pragma unroll
  for (int i=0;i<8;++i) a[i]   = p[kb+i];
#pragma unroll
  for (int i=0;i<8;++i) a[8+i] = p[16+kb+i];
  return a;
}

// Fragment already stored in register order: one 32B vector load per lane.
__device__ __forceinline__ v16h load_frag_packed(const _Float16* base){
  int lane = threadIdx.x & 31;
  return *(const v16h*)(base + (size_t)lane*16);
}

#define WMMA_F16(A,B,Cacc) \
  __builtin_amdgcn_wmma_f32_16x16x32_f16(false,(A),false,(B),(short)0,(Cacc),false,false)

// ---------------- K0: weights fp32 -> f16, pre-swizzled into B-fragment order ---
// WF layout per matrix: [kblk(16)][ntile(32)][lane(32)][elem(16)]
__global__ __launch_bounds__(256) void k_pack_weights(const float* __restrict__ Wq,
    const float* __restrict__ Wk, const float* __restrict__ Wv,
    const float* __restrict__ Wo, _Float16* __restrict__ WF){
  int i = blockIdx.x*256 + threadIdx.x;       // element (k, n) of a 512x512 matrix
  if (i >= CC*CC) return;
  int k = i >> 9, n = i & 511;
  int kblk = k >> 5, kd = k & 31;
  int nt = n >> 4;
  int lane, e; frag_pos(kd, n, lane, e);
  size_t o = (((size_t)kblk*32 + nt)*32 + lane)*16 + e;
  WF[o]                    = (_Float16)Wq[i];
  WF[(size_t)CC*CC   + o]  = (_Float16)Wk[i];
  WF[(size_t)2*CC*CC + o]  = (_Float16)Wv[i];
  WF[(size_t)3*CC*CC + o]  = (_Float16)Wo[i];
}

// ---------------- K1: fused QKV GEMM, sigmoid(q,k), scatter to (B,H,N,D) f16 ----
__global__ __launch_bounds__(256) void k_qkv_gemm(const float* __restrict__ X,
    const _Float16* __restrict__ WF,
    const float* __restrict__ bq, const float* __restrict__ bk, const float* __restrict__ bv,
    _Float16* __restrict__ Qp, _Float16* __restrict__ Kp, _Float16* __restrict__ Vp){
  __shared__ alignas(32) _Float16 As[128*40];   // 128x32 (+8 pad), row-major
  int tid  = threadIdx.x;
  int m0   = blockIdx.x * 128;
  int nblk = blockIdx.y * 128;           // 0..1535 across [q|k|v]
  int sel  = nblk >> 9;                  // 0=q 1=k 2=v (never straddles)
  int n0   = nblk & 511;
  const _Float16* W = WF + (size_t)sel*CC*CC;
  int wave = tid >> 5, mw = wave & 1, nw = wave >> 1;   // waves: 2(M) x 4(N)
  int lane = tid & 31;
  v8f zero = {};
  v8f acc[4][2];
#pragma unroll
  for (int m=0;m<4;++m){ acc[m][0]=zero; acc[m][1]=zero; }

  for (int k0 = 0; k0 < CC; k0 += 32){
    // stage A: 128x32 fp32 -> f16 (float4 loads, 8B LDS stores)
    for (int i = tid; i < 128*32/4; i += 256){
      int idx4 = i*4;
      int r = idx4 >> 5, c = idx4 & 31;
      v4f x = *(const v4f*)&X[(size_t)(m0+r)*CC + k0 + c];
      v4h h; h[0]=(_Float16)x[0]; h[1]=(_Float16)x[1]; h[2]=(_Float16)x[2]; h[3]=(_Float16)x[3];
      *(v4h*)&As[r*40+c] = h;
    }
    __syncthreads();
    const _Float16* wbase = W + ((size_t)(k0>>5)*32)*32*16;
    v16h bfr[2];
#pragma unroll
    for (int n=0;n<2;++n)
      bfr[n] = load_frag_packed(wbase + (size_t)((n0>>4) + nw*2 + n)*32*16);
#pragma unroll
    for (int m=0;m<4;++m){
      v16h afr = load_frag_a(&As[(mw*64 + m*16)*40], 40);
#pragma unroll
      for (int n=0;n<2;++n) acc[m][n] = WMMA_F16(afr, bfr[n], acc[m][n]);
    }
    __syncthreads();
  }

  const float* bias = (sel==0)?bq:((sel==1)?bk:bv);
  _Float16* dst = (sel==0)?Qp:((sel==1)?Kp:Vp);
  int hi = (lane & 16) ? 8 : 0, lc = lane & 15;
#pragma unroll
  for (int m=0;m<4;++m)
#pragma unroll
  for (int n=0;n<2;++n)
#pragma unroll
  for (int j=0;j<8;++j){
    int row = m0 + mw*64 + m*16 + j + hi;   // = nseq*B + b
    int col = n0 + nw*32 + n*16 + lc;       // 0..511 = h*64+d
    float v = acc[m][n][j] + bias[col];
    if (sel < 2) v = sigmoidf(v);
    int nn = row >> 3, b = row & 7;
    int h  = col >> 6, d = col & 63;
    dst[(size_t)((b*HH + h)*NSEQ + nn)*DD + d] = (_Float16)v;
  }
}

// ---------------- K2/K4: per-(b,h,d) (weighted) column sums over n --------------
// thread t: d-chunk (t&7)*8..+8 (one 16B v8h load), n-lane t>>3 of 32
__global__ __launch_bounds__(256) void k_colsum(const _Float16* __restrict__ Kp,
    const _Float16* __restrict__ Qp, const float* __restrict__ wk,
    const float* __restrict__ wq, float* __restrict__ outK, float* __restrict__ outQ){
  int bh = blockIdx.x, t = threadIdx.x;
  int dgrp = (t & 7) * 8;
  int nl = t >> 3;
  float s1[8], s2[8];
#pragma unroll
  for (int i=0;i<8;++i){ s1[i]=0.f; s2[i]=0.f; }
  for (int n = nl; n < NSEQ; n += 32){
    size_t g = (size_t)(bh*NSEQ + n)*DD + dgrp;
    v8h kv8 = *(const v8h*)&Kp[g];
    v8h qv8 = *(const v8h*)&Qp[g];
    float w1 = wk ? wk[bh*NSEQ + n] : 1.0f;
    float w2 = wq ? wq[bh*NSEQ + n] : 1.0f;
#pragma unroll
    for (int i=0;i<8;++i){ s1[i] += (float)kv8[i]*w1; s2[i] += (float)qv8[i]*w2; }
  }
  __shared__ float r1[2048], r2[2048];
#pragma unroll
  for (int i=0;i<8;++i){ r1[t*8+i]=s1[i]; r2[t*8+i]=s2[i]; }
  __syncthreads();
  if (t < 64){                      // output d = t
    int i = t & 7, tg = t >> 3;
    float a = 0.f, b = 0.f;
    for (int n2 = 0; n2 < 32; ++n2){
      int slot = (n2*8 + tg)*8 + i;
      a += r1[slot]; b += r2[slot];
    }
    outK[bh*DD + t] = a;
    outQ[bh*DD + t] = b;
  }
}

// ---------------- K3: sink_incoming / source_outgoing ---------------------------
__global__ __launch_bounds__(256) void k_norm(const _Float16* __restrict__ Qp,
    const _Float16* __restrict__ Kp, const float* __restrict__ ksum,
    const float* __restrict__ qsum, float* __restrict__ si, float* __restrict__ so){
  int idx = blockIdx.x*256 + threadIdx.x;   // bh*4096 + n
  int bh  = idx >> 12;
  __shared__ float sk[64], sq[64];
  int t = threadIdx.x;
  if (t < 64) sk[t] = ksum[bh*DD + t];
  else if (t < 128) sq[t-64] = qsum[bh*DD + (t-64)];
  __syncthreads();
  float d1 = 0.f, d2 = 0.f;
  const _Float16* qp = Qp + (size_t)idx*DD;
  const _Float16* kp = Kp + (size_t)idx*DD;
#pragma unroll
  for (int c=0;c<4;++c){
    v16h qv = *(const v16h*)(qp + c*16);
    v16h kv = *(const v16h*)(kp + c*16);
#pragma unroll
    for (int i=0;i<16;++i){
      d1 += ((float)qv[i] + EPSF) * (sk[c*16+i] + EPSF);
      d2 += ((float)kv[i] + EPSF) * (sq[c*16+i] + EPSF);
    }
  }
  si[idx] = fast_rcp(d1 + EPSF);
  so[idx] = fast_rcp(d2 + EPSF);
}

// ---------------- K5: conserved sink/source, allocation, clip -------------------
__global__ __launch_bounds__(256) void k_conserve(const _Float16* __restrict__ Qp,
    const _Float16* __restrict__ Kp, const float* __restrict__ kws,
    const float* __restrict__ qws, float* __restrict__ sa, float* __restrict__ csrc){
  int idx = blockIdx.x*256 + threadIdx.x;
  int bh  = idx >> 12;
  __shared__ float sk[64], sq[64];
  int t = threadIdx.x;
  if (t < 64) sk[t] = kws[bh*DD + t];
  else if (t < 128) sq[t-64] = qws[bh*DD + (t-64)];
  __syncthreads();
  float d1 = 0.f, d2 = 0.f;
  const _Float16* qp = Qp + (size_t)idx*DD;
  const _Float16* kp = Kp + (size_t)idx*DD;
#pragma unroll
  for (int c=0;c<4;++c){
    v16h qv = *(const v16h*)(qp + c*16);
    v16h kv = *(const v16h*)(kp + c*16);
#pragma unroll
    for (int i=0;i<16;++i){
      d1 += ((float)qv[i] + EPSF) * (sk[c*16+i] + EPSF);
      d2 += ((float)kv[i] + EPSF) * (sq[c*16+i] + EPSF);
    }
  }
  float csink = d1 + EPSF;
  sa[idx] = sigmoidf(csink * ((float)NSEQ/(float)NSEQ));   // n/hw == 1 here
  float cs = d2 + EPSF;
  csrc[idx] = fminf(1.0f, fmaxf(-1.0f, cs));
}

// ---------------- K6: softmax over key axis (4096) x C --------------------------
__global__ __launch_bounds__(256) void k_softmax(const float* __restrict__ csrc,
    const float* __restrict__ temp, float* __restrict__ comp){
  int bh = blockIdx.x, t = threadIdx.x;
  float invT = fast_rcp(temp[0]);
  __shared__ float red[256];
  float m = -3.4e38f;
  for (int n=t; n<NSEQ; n+=256) m = fmaxf(m, csrc[bh*NSEQ+n]*invT);
  red[t]=m; __syncthreads();
  for (int s=128; s>0; s>>=1){ if (t<s) red[t]=fmaxf(red[t],red[t+s]); __syncthreads(); }
  float mx = red[0]; __syncthreads();
  float s = 0.f;
  for (int n=t; n<NSEQ; n+=256) s += __expf(csrc[bh*NSEQ+n]*invT - mx);
  red[t]=s; __syncthreads();
  for (int st=128; st>0; st>>=1){ if (t<st) red[t]+=red[t+st]; __syncthreads(); }
  float scale = (float)CC * fast_rcp(red[0]);
  for (int n=t; n<NSEQ; n+=256)
    comp[bh*NSEQ+n] = __expf(csrc[bh*NSEQ+n]*invT - mx)*scale;
}

// ---------------- K7: kv[bh] = k^T (DxN) x (v*comp) (NxD) via WMMA --------------
// LDS tiles staged directly in WMMA fragment order (vector global loads,
// scattered b16 LDS stores for the transpose; contiguous b128 on the hot read
// side). Result written to kvF in B-fragment order for K8:
// kvF layout [bh][ks(2)][ntile(4)][lane(32)][elem(16)].
__global__ __launch_bounds__(256) void k_kv(const _Float16* __restrict__ Kp,
    const _Float16* __restrict__ Vp, const float* __restrict__ comp,
    _Float16* __restrict__ kvF){
  int bh = blockIdx.x, tid = threadIdx.x;
  __shared__ alignas(32) _Float16 KsF[4*32*16];  // A-frags (k^T) for 4 m-tiles
  __shared__ alignas(32) _Float16 VsF[4*32*16];  // B-frags (v*comp) for 4 n-tiles
  int wave = tid >> 5;
  int mt = wave >> 1;          // d-row tile 0..3
  int nb = (wave & 1)*2;       // d-col tiles {0,1} or {2,3}
  int r  = tid >> 3;           // staging: local n (K dim) 0..31
  int c0 = (tid & 7) * 8;      // staging: d base (one v8h per thread)
  int e  = (r & 7) + ((r & 16) ? 8 : 0);     // frag elem index (depends on r only)
  int lhi = (r & 8) ? 16 : 0;                // lane-half from K index
  v8f zero = {};
  v8f acc[2]; acc[0]=zero; acc[1]=zero;
  for (int n0=0; n0<NSEQ; n0+=32){
    size_t g = (size_t)(bh*NSEQ + n0 + r)*DD + c0;
    v8h kv8 = *(const v8h*)&Kp[g];
    v8h vv8 = *(const v8h*)&Vp[g];
    float w = comp[bh*NSEQ + n0 + r];
#pragma unroll
    for (int i=0;i<8;++i){
      int c = c0 + i;
      size_t o = ((size_t)(c>>4)*32 + ((c & 15) + lhi))*16 + e;
      KsF[o] = kv8[i];
      VsF[o] = (_Float16)((float)vv8[i]*w);
    }
    __syncthreads();
    v16h afr = load_frag_packed(&KsF[(size_t)mt*32*16]);
#pragma unroll
    for (int n=0;n<2;++n){
      v16h bfr = load_frag_packed(&VsF[(size_t)(nb+n)*32*16]);
      acc[n] = WMMA_F16(afr, bfr, acc[n]);
    }
    __syncthreads();
  }
  int lane = tid & 31;
  int hi = (lane&16)?8:0, lc = lane & 15;
#pragma unroll
  for (int n=0;n<2;++n)
#pragma unroll
  for (int j=0;j<8;++j){
    int drow = mt*16 + j + hi;        // K index of the next GEMM
    int ecol = (nb+n)*16 + lc;        // N index of the next GEMM
    int ks = drow >> 5, kd = drow & 31;
    int l2, e2; frag_pos(kd, ecol, l2, e2);
    kvF[(((size_t)(bh*2 + ks)*4 + (ecol>>4))*32 + l2)*16 + e2] = (_Float16)acc[n][j];
  }
}

// ---------------- K8: out_update = q @ kv, scale, scatter to (B,N,C) f16 --------
__global__ __launch_bounds__(256) void k_out_update(const _Float16* __restrict__ Qp,
    const _Float16* __restrict__ kvF, const float* __restrict__ si,
    const float* __restrict__ sa, _Float16* __restrict__ Uo){
  int bh = blockIdx.x;
  int b = bh >> 3, h = bh & 7;
  int n0 = blockIdx.y * 128;
  int wave = threadIdx.x >> 5;
  int nrow = n0 + wave*16;
  const _Float16* Abase = Qp + (size_t)(bh*NSEQ + nrow)*DD;
  const _Float16* Bbase = kvF + (size_t)bh*2*4*32*16;
  v8f zero = {};
  v8f acc[4];
#pragma unroll
  for (int n=0;n<4;++n) acc[n]=zero;
#pragma unroll
  for (int ks=0; ks<2; ++ks){
    v16h afr = load_frag_a(Abase + ks*32, DD);
#pragma unroll
    for (int n=0;n<4;++n){
      v16h bfr = load_frag_packed(Bbase + (size_t)(ks*4 + n)*32*16);
      acc[n] = WMMA_F16(afr, bfr, acc[n]);
    }
  }
  int lane = threadIdx.x & 31;
  int hi = (lane&16)?8:0, lc = lane & 15;
#pragma unroll
  for (int n=0;n<4;++n)
#pragma unroll
  for (int j=0;j<8;++j){
    int nn  = nrow + j + hi;
    int idx = bh*NSEQ + nn;
    float v = acc[n][j] * si[idx] * sa[idx];
    int d  = n*16 + lc;
    int np = h*512 + (nn >> 3);            // (B,H,N,D) -> (B,N,C) reshape
    int cp = ((nn & 7) << 6) + d;
    Uo[(size_t)(b*NSEQ + np)*CC + cp] = (_Float16)v;
  }
}

// ---------------- K9: Out = Uo @ Wo + bo + residual, write (N,B,C) fp32 ---------
__global__ __launch_bounds__(256) void k_out_gemm(const _Float16* __restrict__ Uo,
    const _Float16* __restrict__ WF, const float* __restrict__ bo,
    const float* __restrict__ query, float* __restrict__ out){
  __shared__ alignas(32) _Float16 As[128*40];
  const _Float16* W = WF + (size_t)3*CC*CC;
  int tid = threadIdx.x;
  int m0 = blockIdx.x*128, n0 = blockIdx.y*128;
  int wave = tid>>5, mw = wave&1, nw = wave>>1;
  int lane = tid & 31;
  v8f zero = {};
  v8f acc[4][2];
#pragma unroll
  for (int m=0;m<4;++m){ acc[m][0]=zero; acc[m][1]=zero; }
  for (int k0=0;k0<CC;k0+=32){
    // stage A: 128x32 f16 copy (16B vector loads/stores)
    for (int i=tid; i<128*32/8; i+=256){
      int idx8 = i*8;
      int r = idx8 >> 5, c = idx8 & 31;
      *(v8h*)&As[r*40+c] = *(const v8h*)&Uo[(size_t)(m0+r)*CC + k0+c];
    }
    __syncthreads();
    const _Float16* wbase = W + ((size_t)(k0>>5)*32)*32*16;
    v16h bfr[2];
#pragma unroll
    for (int n=0;n<2;++n)
      bfr[n] = load_frag_packed(wbase + (size_t)((n0>>4) + nw*2 + n)*32*16);
#pragma unroll
    for (int m=0;m<4;++m){
      v16h afr = load_frag_a(&As[(mw*64 + m*16)*40], 40);
#pragma unroll
      for (int n=0;n<2;++n) acc[m][n] = WMMA_F16(afr, bfr[n], acc[m][n]);
    }
    __syncthreads();
  }
  int hi = (lane&16)?8:0, lc = lane & 15;
#pragma unroll
  for (int m=0;m<4;++m)
#pragma unroll
  for (int n=0;n<2;++n)
#pragma unroll
  for (int j=0;j<8;++j){
    int row = m0 + mw*64 + m*16 + j + hi;    // = b*4096 + n'
    int col = n0 + nw*32 + n*16 + lc;
    int b = row >> 12, np = row & 4095;
    size_t o = (size_t)(np*BB + b)*CC + col; // (N,B,C)
    out[o] = query[o] + acc[m][n][j] + bo[col];
  }
}

extern "C" void kernel_launch(void* const* d_in, const int* in_sizes, int n_in,
                              void* d_out, int out_size, void* d_ws, size_t ws_size,
                              hipStream_t stream) {
  (void)in_sizes; (void)n_in; (void)out_size; (void)ws_size;
  const float* X    = (const float*)d_in[0];
  const float* Wq   = (const float*)d_in[1];
  const float* bq   = (const float*)d_in[2];
  const float* Wk   = (const float*)d_in[3];
  const float* bk   = (const float*)d_in[4];
  const float* Wv   = (const float*)d_in[5];
  const float* bv   = (const float*)d_in[6];
  const float* Wo   = (const float*)d_in[7];
  const float* bo   = (const float*)d_in[8];
  const float* temp = (const float*)d_in[9];
  float* out = (float*)d_out;

  char* ws = (char*)d_ws;
  size_t off = 0;
  auto alloc = [&](size_t bytes)->void*{
    void* p = ws + off; off += (bytes + 255) & ~(size_t)255; return p;
  };
  _Float16* WF   = (_Float16*)alloc((size_t)4*CC*CC*2);        //  2 MB (fragment order)
  _Float16* Qp   = (_Float16*)alloc((size_t)BH*NSEQ*DD*2);     // 32 MB (B,H,N,D)
  _Float16* Kp   = (_Float16*)alloc((size_t)BH*NSEQ*DD*2);     // 32 MB
  _Float16* Vp   = (_Float16*)alloc((size_t)BH*NSEQ*DD*2);     // 32 MB
  _Float16* Uo   = (_Float16*)alloc((size_t)BB*NSEQ*CC*2);     // 32 MB (B,N,C)
  _Float16* kvF  = (_Float16*)alloc((size_t)BH*DD*DD*2);       // 0.5 MB (fragment order)
  float* ksum = (float*)alloc((size_t)BH*DD*4);
  float* qsum = (float*)alloc((size_t)BH*DD*4);
  float* kws  = (float*)alloc((size_t)BH*DD*4);
  float* qws  = (float*)alloc((size_t)BH*DD*4);
  float* si   = (float*)alloc((size_t)BH*NSEQ*4);              // 1 MB each
  float* so   = (float*)alloc((size_t)BH*NSEQ*4);
  float* sa   = (float*)alloc((size_t)BH*NSEQ*4);
  float* csrc = (float*)alloc((size_t)BH*NSEQ*4);
  float* comp = (float*)alloc((size_t)BH*NSEQ*4);

  dim3 blk(256);
  k_pack_weights<<<dim3((CC*CC+255)/256), blk, 0, stream>>>(Wq,Wk,Wv,Wo,WF);
  k_qkv_gemm  <<<dim3(MTOT/128, (3*CC)/128), blk, 0, stream>>>(X,WF,bq,bk,bv,Qp,Kp,Vp);
  k_colsum    <<<dim3(BH), blk, 0, stream>>>(Kp,Qp,(const float*)nullptr,(const float*)nullptr,ksum,qsum);
  k_norm      <<<dim3((BH*NSEQ)/256), blk, 0, stream>>>(Qp,Kp,ksum,qsum,si,so);
  k_colsum    <<<dim3(BH), blk, 0, stream>>>(Kp,Qp,so,si,kws,qws);
  k_conserve  <<<dim3((BH*NSEQ)/256), blk, 0, stream>>>(Qp,Kp,kws,qws,sa,csrc);
  k_softmax   <<<dim3(BH), blk, 0, stream>>>(csrc,temp,comp);
  k_kv        <<<dim3(BH), blk, 0, stream>>>(Kp,Vp,comp,kvF);
  k_out_update<<<dim3(BH, NSEQ/128), blk, 0, stream>>>(Qp,kvF,si,sa,Uo);
  k_out_gemm  <<<dim3(MTOT/128, CC/128), blk, 0, stream>>>(Uo,WF,bo,X,out);
}
